// Net_30107720745841
// MI455X (gfx1250) — matrix operand order repaired and checked by reading, not compiled, not run on hardware
//
#include <hip/hip_runtime.h>

typedef __attribute__((ext_vector_type(16))) _Float16 v16h;
typedef __attribute__((ext_vector_type(8)))  _Float16 v8h;
typedef __attribute__((ext_vector_type(8)))  float    v8f;

#define N_NODES 100000
#define N_EDGES 1600000
#define N_LABEL 200000
#define IN_C  64
#define HID_C 128
#define OUT_C 64

// ---------------- zero a float range ----------------
__global__ void zero_kernel(float* __restrict__ p, long n) {
  long i = (long)blockIdx.x * blockDim.x + threadIdx.x;
  long stride = (long)gridDim.x * blockDim.x;
  for (; i < n; i += stride) p[i] = 0.0f;
}

// ---------------- f32 -> f16 elementwise convert ----------------
__global__ void cvt_f16_kernel(const float* __restrict__ in,
                               _Float16* __restrict__ out, long n) {
  long i = (long)blockIdx.x * blockDim.x + threadIdx.x;
  long stride = (long)gridDim.x * blockDim.x;
  for (; i < n; i += stride) out[i] = (_Float16)in[i];
}

// ---------------- W[K][N] f32 -> WT[N][K] f16 (transpose for B fragments) ----------------
__global__ void transpose_w_kernel(const float* __restrict__ W,
                                   _Float16* __restrict__ WT, int K, int N) {
  int i = blockIdx.x * blockDim.x + threadIdx.x;
  if (i >= K * N) return;
  int n = i / K, k = i % K;
  WT[(size_t)n * K + k] = (_Float16)W[(size_t)k * N + n];
}

// ---------------- agg(f32) * 1/max(cnt,1) -> f16 (mean fold-in) ----------------
__global__ void scale_cvt_kernel(const float* __restrict__ agg,
                                 const float* __restrict__ cnt,
                                 _Float16* __restrict__ out, int C, long n) {
  long i = (long)blockIdx.x * blockDim.x + threadIdx.x;
  long stride = (long)gridDim.x * blockDim.x;
  for (; i < n; i += stride) {
    int row = (int)(i / C);
    float inv = 1.0f / fmaxf(cnt[row], 1.0f);
    out[i] = (_Float16)(agg[i] * inv);
  }
}

// ---------------- edge-parallel scatter-add ----------------
__global__ void scatter_accum(const float* __restrict__ feat,
                              const int* __restrict__ src,
                              const int* __restrict__ dst,
                              float* __restrict__ agg,
                              float* __restrict__ cnt,
                              int nE, int C) {
  const int c4 = C >> 2;
  const long total = (long)nE * c4;
  long i = (long)blockIdx.x * blockDim.x + threadIdx.x;
  const long stride = (long)gridDim.x * blockDim.x;
  for (; i < total; i += stride) {
    int e = (int)(i / c4);
    int c = (int)(i % c4) * 4;
    int s = src[e], d = dst[e];
    const float4 v = *(const float4*)(feat + (size_t)s * C + c);
    float* out = agg + (size_t)d * C + c;
    atomicAdd(out + 0, v.x);
    atomicAdd(out + 1, v.y);
    atomicAdd(out + 2, v.z);
    atomicAdd(out + 3, v.w);
    if (cnt != nullptr && c == 0) atomicAdd(cnt + d, 1.0f);
  }
}

// ---------------- WMMA fragment loads: pure contiguous f16 vector loads ----------------
// A 16x32 f16 (row-major, row stride = KD halfs): lane(half,lm) row=lm,
//   VGPR0-3: K = k0 + half*16 + [0,8)   -> one 16B chunk
//   VGPR4-7: K = k0 + half*16 + 8 + [0,8)-> one 16B chunk
__device__ __forceinline__ v16h load_a16(const _Float16* __restrict__ row,
                                         int k0, int half) {
  v8h lo = *(const v8h*)(row + k0 + half * 16);
  v8h hi = *(const v8h*)(row + k0 + half * 16 + 8);
  v16h a;
#pragma unroll
  for (int i = 0; i < 8; ++i) { a[i] = lo[i]; a[8 + i] = hi[i]; }
  return a;
}

// B 32x16 f16 from col-major WT[col][K]: lane(half,lm) col=lm,
//   VGPR j: K = k0 + half*16 + 2j + p  -> one contiguous 32B run
__device__ __forceinline__ v16h load_b16(const _Float16* __restrict__ colRow,
                                         int k0, int half) {
  return *(const v16h*)(colRow + k0 + half * 16);
}

// ---------------- SAGEConv GEMM: out = meanAgg@Wl + x@Wr + bias ----------------
// Block = 16-node tile; wave w owns output columns [16w,16w+16).
template<int KD, int ND, bool RELU, bool DUAL>
__global__ void sage_gemm(const _Float16* __restrict__ aggH,  // pre-scaled mean, f16
                          const _Float16* __restrict__ xH,    // self features, f16
                          const _Float16* __restrict__ WlT,   // [ND][KD] f16
                          const _Float16* __restrict__ WrT,   // [ND][KD] f16
                          const float*    __restrict__ bias,
                          float*          __restrict__ outF,
                          _Float16*       __restrict__ outH) {
  const int tile = blockIdx.x;
  const int wave = threadIdx.x >> 5;
  const int lane = threadIdx.x & 31;
  const int half = lane >> 4;
  const int lm   = lane & 15;
  const int row  = tile * 16 + lm;
  const int col  = wave * 16 + lm;

  const _Float16* aRow  = aggH + (size_t)row * KD;
  const _Float16* xRow  = xH   + (size_t)row * KD;
  const _Float16* blCol = WlT  + (size_t)col * KD;
  const _Float16* brCol = WrT  + (size_t)col * KD;

  v8f acc = {};
#pragma unroll
  for (int k0 = 0; k0 < KD; k0 += 32) {
    v16h aA = load_a16(aRow, k0, half);
    v16h bA = load_b16(blCol, k0, half);
    acc = __builtin_amdgcn_wmma_f32_16x16x32_f16(false, aA, false, bA,
                                                 (short)0, acc, false, false);
    v16h aX = load_a16(xRow, k0, half);
    v16h bX = load_b16(brCol, k0, half);
    acc = __builtin_amdgcn_wmma_f32_16x16x32_f16(false, aX, false, bX,
                                                 (short)0, acc, false, false);
  }

  const float bb = bias[col];
#pragma unroll
  for (int r = 0; r < 8; ++r) {
    int m = tile * 16 + half * 8 + r;       // C/D layout: M = r + 8*half, N = lm
    float v = acc[r] + bb;
    if (RELU) v = fmaxf(v, 0.0f);
    outF[(size_t)m * ND + col] = v;
    if (DUAL) outH[(size_t)m * ND + col] = (_Float16)v;
  }
}

// ---------------- decode: logits[p] = dot(z[i], z[j]) ----------------
__global__ void decode_kernel(const float* __restrict__ z,
                              const int* __restrict__ li,
                              const int* __restrict__ lj,
                              float* __restrict__ out, int nP) {
  int p = blockIdx.x * blockDim.x + threadIdx.x;
  if (p >= nP) return;
  const float4* a = (const float4*)(z + (size_t)li[p] * OUT_C);
  const float4* b = (const float4*)(z + (size_t)lj[p] * OUT_C);
  float s = 0.0f;
#pragma unroll
  for (int q = 0; q < OUT_C / 4; ++q) {
    float4 va = a[q], vb = b[q];
    s += va.x * vb.x + va.y * vb.y + va.z * vb.z + va.w * vb.w;
  }
  out[p] = s;
}

extern "C" void kernel_launch(void* const* d_in, const int* in_sizes, int n_in,
                              void* d_out, int out_size, void* d_ws, size_t ws_size,
                              hipStream_t stream) {
  (void)in_sizes; (void)n_in; (void)out_size; (void)ws_size;

  const float* x   = (const float*)d_in[0];
  const int*   ei  = (const int*)d_in[1];     // [2, N_EDGES]
  const int*   eli = (const int*)d_in[2];     // [2, N_LABEL]
  const float* Wl1 = (const float*)d_in[3];   // [64,128]
  const float* Wr1 = (const float*)d_in[4];   // [64,128]
  const float* b1  = (const float*)d_in[5];   // [128]
  const float* Wl2 = (const float*)d_in[6];   // [128,64]
  const float* Wr2 = (const float*)d_in[7];   // [128,64]
  const float* b2  = (const float*)d_in[8];   // [64]
  float* logits = (float*)d_out;

  // -------- workspace layout --------
  // f32 region: cnt | agg1 | agg2 | h | z
  float* wsf  = (float*)d_ws;
  float* cnt  = wsf;
  float* agg1 = cnt  + N_NODES;                      // N*64
  float* agg2 = agg1 + (size_t)N_NODES * IN_C;       // N*128
  float* h    = agg2 + (size_t)N_NODES * HID_C;      // N*128
  float* z    = h    + (size_t)N_NODES * HID_C;      // N*64
  // f16 region: xh | agg1h | hh | agg2h | WlT1 | WrT1 | WlT2 | WrT2
  _Float16* wsh   = (_Float16*)(z + (size_t)N_NODES * OUT_C);
  _Float16* xh    = wsh;
  _Float16* agg1h = xh    + (size_t)N_NODES * IN_C;
  _Float16* hh    = agg1h + (size_t)N_NODES * IN_C;
  _Float16* agg2h = hh    + (size_t)N_NODES * HID_C;
  _Float16* WlT1  = agg2h + (size_t)N_NODES * HID_C; // [128][64]
  _Float16* WrT1  = WlT1 + IN_C * HID_C;
  _Float16* WlT2  = WrT1 + IN_C * HID_C;             // [64][128]
  _Float16* WrT2  = WlT2 + HID_C * OUT_C;

  const int* src = ei;
  const int* dst = ei + N_EDGES;

  // 1) zero cnt + agg1 + agg2 (contiguous prefix)
  long nz = (long)N_NODES + (long)N_NODES * IN_C + (long)N_NODES * HID_C;
  zero_kernel<<<2048, 256, 0, stream>>>(wsf, nz);

  // 2) stage f16 operands: x and transposed weights
  cvt_f16_kernel<<<2048, 256, 0, stream>>>(x, xh, (long)N_NODES * IN_C);
  transpose_w_kernel<<<(IN_C * HID_C + 255) / 256, 256, 0, stream>>>(Wl1, WlT1, IN_C, HID_C);
  transpose_w_kernel<<<(IN_C * HID_C + 255) / 256, 256, 0, stream>>>(Wr1, WrT1, IN_C, HID_C);
  transpose_w_kernel<<<(HID_C * OUT_C + 255) / 256, 256, 0, stream>>>(Wl2, WlT2, HID_C, OUT_C);
  transpose_w_kernel<<<(HID_C * OUT_C + 255) / 256, 256, 0, stream>>>(Wr2, WrT2, HID_C, OUT_C);

  // 3) layer-1 neighbor sum + degree count, then mean+convert to f16
  scatter_accum<<<4096, 256, 0, stream>>>(x, src, dst, agg1, cnt, N_EDGES, IN_C);
  scale_cvt_kernel<<<2048, 256, 0, stream>>>(agg1, cnt, agg1h, IN_C,
                                             (long)N_NODES * IN_C);

  // 4) h = relu(mean @ Wl1 + x @ Wr1 + b1)  [WMMA]; dual-store f32 + f16
  sage_gemm<IN_C, HID_C, true, true><<<N_NODES / 16, 256, 0, stream>>>(
      agg1h, xh, WlT1, WrT1, b1, h, hh);

  // 5) layer-2 neighbor sum, then mean+convert to f16
  scatter_accum<<<8192, 256, 0, stream>>>(h, src, dst, agg2, nullptr, N_EDGES, HID_C);
  scale_cvt_kernel<<<2048, 256, 0, stream>>>(agg2, cnt, agg2h, HID_C,
                                             (long)N_NODES * HID_C);

  // 6) z = mean @ Wl2 + h @ Wr2 + b2        [WMMA]
  sage_gemm<HID_C, OUT_C, false, false><<<N_NODES / 16, 128, 0, stream>>>(
      agg2h, hh, WlT2, WrT2, b2, z, nullptr);

  // 7) decode
  decode_kernel<<<(N_LABEL + 255) / 256, 256, 0, stream>>>(
      z, eli, eli + N_LABEL, logits, N_LABEL);
}
